// LSTMClassifier_71356586656121
// MI455X (gfx1250) — compile-verified, matrix operand order checked by reading, not run on hardware
//
#include <hip/hip_runtime.h>
#include <hip/hip_bf16.h>
#include <math.h>

// ---------------- problem constants ----------------
#define BATCH  64
#define TLEN   512
#define INDIM  128
#define HDIM   512
#define G4H    (4 * HDIM)      // 2048 gate rows
#define NWG    32              // workgroups in the scan (H / HC)
#define HC     16              // hidden columns owned per workgroup
#define SCAN_THREADS 256       // 8 waves; waves 0..3 do WMMA, all stage
#define LDS_PAD 8              // bf16 elements of row padding (bank spread)

typedef __bf16 bf16_t;
typedef __attribute__((ext_vector_type(16))) __bf16 v16bf;
typedef __attribute__((ext_vector_type(8)))  __bf16 v8bf;
typedef __attribute__((ext_vector_type(8)))  float  v8f;

// ---------------- fast branch-free activations ----------------
// sigmoid(x) = 1/(1+e^-x); e^-x=inf -> 0, e^-x=0 -> 1 (saturates correctly)
__device__ __forceinline__ float sigmoidf_(float x) {
    return __builtin_amdgcn_rcpf(1.0f + __expf(-x));
}
// tanh(x) = 1 - 2/(e^{2x}+1); e^{2x}=inf -> 1, e^{2x}=0 -> -1
__device__ __forceinline__ float tanhf_(float x) {
    return 1.0f - 2.0f * __builtin_amdgcn_rcpf(__expf(2.0f * x) + 1.0f);
}

// Build a WMMA 16-element bf16 fragment from two contiguous 8-element chunks.
// Vector elements 0..7 = K {kh..kh+7}, elements 8..15 = K {16+kh..23+kh}
// (matches CDNA5 16-bit A/B VGPR packing for wave32).
__device__ __forceinline__ v16bf ld_frag(const bf16_t* p0, const bf16_t* p1) {
    v8bf lo = *(const v8bf*)p0;
    v8bf hi = *(const v8bf*)p1;
    v16bf r;
#pragma unroll
    for (int i = 0; i < 8; ++i) { r[i] = lo[i]; r[i + 8] = hi[i]; }
    return r;
}

// Step-indexed global barrier across NWG persistent workgroups.
__device__ __forceinline__ void grid_barrier(int* cnt, int idx) {
    __threadfence();           // make this WG's global stores visible (agent)
    __syncthreads();
    if (threadIdx.x == 0) {
        __hip_atomic_fetch_add(&cnt[idx], 1, __ATOMIC_RELEASE,
                               __HIP_MEMORY_SCOPE_AGENT);
        while (__hip_atomic_load(&cnt[idx], __ATOMIC_ACQUIRE,
                                 __HIP_MEMORY_SCOPE_AGENT) < NWG) {
            __builtin_amdgcn_s_sleep(2);
        }
    }
    __syncthreads();
    __threadfence();           // acquire side: invalidate stale lines
}

// ---------------- precompute kernels ----------------
__global__ void f32_to_bf16_kernel(const float* __restrict__ in,
                                   bf16_t* __restrict__ out, int n) {
    for (int i = blockIdx.x * blockDim.x + threadIdx.x; i < n;
         i += gridDim.x * blockDim.x)
        out[i] = (bf16_t)in[i];
}

// Pack [w_ih | w_hh] row-major into bf16 [4H][Kin+H].
__global__ void pack_w_kernel(const float* __restrict__ w_ih,
                              const float* __restrict__ w_hh,
                              bf16_t* __restrict__ out, int Kin) {
    const int K = Kin + HDIM;
    const int total = G4H * K;
    for (int i = blockIdx.x * blockDim.x + threadIdx.x; i < total;
         i += gridDim.x * blockDim.x) {
        int r = i / K, k = i % K;
        float v = (k < Kin) ? w_ih[r * Kin + k] : w_hh[r * HDIM + (k - Kin)];
        out[i] = (bf16_t)v;
    }
}

__global__ void bias_kernel(const float* __restrict__ b_ih,
                            const float* __restrict__ b_hh,
                            float* __restrict__ out) {
    int i = blockIdx.x * blockDim.x + threadIdx.x;
    if (i < G4H) out[i] = b_ih[i] + b_hh[i];
}

__global__ void reset_kernel(int* cnt, int n) {
    int i = blockIdx.x * blockDim.x + threadIdx.x;
    if (i < n) cnt[i] = 0;
}

// ---------------- fused LSTM layer scan (persistent, WMMA) ----------------
// KIN is compile-time so K/LDSK fold into immediate ds_load offsets.
// xseq  : [B][T][KIN] bf16 (layer input sequence)
// Wpack : [4H][K] bf16 where K = KIN + H, rows = gate order i,f,g,o
// bias  : [4H] f32 (b_ih + b_hh)
// hcur  : [B][H] bf16 (working hidden state, zero-initialized in-kernel)
// outseq: [B][T][H] bf16 (this layer's hidden sequence)
// cnt   : [T+1] ints, pre-zeroed by reset_kernel
template <int KIN>
__global__ __launch_bounds__(SCAN_THREADS)
void lstm_scan_kernel(const bf16_t* __restrict__ xseq,
                      const bf16_t* __restrict__ Wpack,
                      const float* __restrict__ bias,
                      bf16_t* hcur, bf16_t* __restrict__ outseq, int* cnt) {
    constexpr int K    = KIN + HDIM;
    constexpr int LDSK = K + LDS_PAD;

    extern __shared__ char smem_raw[];
    bf16_t* Wlds = (bf16_t*)smem_raw;        // [64][LDSK]
    bf16_t* xh   = Wlds + 64 * LDSK;         // [64][LDSK]  ([x_t | h_{t-1}])

    const int tid    = threadIdx.x;
    const int wgbase = blockIdx.x * HC;      // this WG's hidden column base
    const int lane   = tid & 31;
    const int wave   = tid >> 5;             // waves 0..3 compute
    const int nn     = lane & 15;            // N col / M row within tile
    const int kh     = (lane >> 4) << 3;     // 0 or 8: K-half, also C-row off

    // ---- load this WG's 64 gate rows of Wpack into LDS (once) ----
    {
        constexpr int cpr = K / 8;           // 16B chunks per row
        for (int c = tid; c < 64 * cpr; c += SCAN_THREADS) {
            int lr = c / cpr;                // local row: g*16 + r
            int kc = (c % cpr) * 8;
            int grow = (lr >> 4) * HDIM + wgbase + (lr & 15);
            *(v8bf*)&Wlds[lr * LDSK + kc] =
                *(const v8bf*)&Wpack[(size_t)grow * K + kc];
        }
    }

    // ---- zero our slice of hcur (h_{-1} = 0) ----
    for (int i = tid; i < BATCH * HC; i += SCAN_THREADS) {
        int m = i / HC, j = i % HC;
        hcur[m * HDIM + wgbase + j] = (bf16_t)0.0f;
    }

    // ---- per-lane biases and persistent cell state (compute waves) ----
    float bi = 0.f, bff = 0.f, bg = 0.f, bo = 0.f;
    if (wave < 4) {
        bi  = bias[0 * HDIM + wgbase + nn];
        bff = bias[1 * HDIM + wgbase + nn];
        bg  = bias[2 * HDIM + wgbase + nn];
        bo  = bias[3 * HDIM + wgbase + nn];
    }
    v8f creg = {0.f, 0.f, 0.f, 0.f, 0.f, 0.f, 0.f, 0.f};

    grid_barrier(cnt, 0);                    // W/h init visible everywhere

    for (int t = 0; t < TLEN; ++t) {
        // ---- stage [x_t | h_{t-1}] into LDS (all 8 waves) ----
        constexpr int xcpr = KIN / 8;
        for (int c = tid; c < BATCH * xcpr; c += SCAN_THREADS) {
            int m = c / xcpr, kc = (c % xcpr) * 8;
            *(v8bf*)&xh[m * LDSK + kc] =
                *(const v8bf*)&xseq[((size_t)m * TLEN + t) * KIN + kc];
        }
        constexpr int hcpr = HDIM / 8;
        for (int c = tid; c < BATCH * hcpr; c += SCAN_THREADS) {
            int m = c / hcpr, kc = (c % hcpr) * 8;
            *(v8bf*)&xh[m * LDSK + KIN + kc] =
                *(const v8bf*)&hcur[m * HDIM + kc];
        }
        __syncthreads();

        if (wave < 4) {
            // gates[M-tile=wave][i,f,g,o] = [x_t|h] @ Wslice^T  (bf16->f32)
            v8f a0 = {0,0,0,0,0,0,0,0}, a1 = a0, a2 = a0, a3 = a0;
            const bf16_t* arow = &xh[(wave * 16 + nn) * LDSK + kh];
            const bf16_t* b0r  = &Wlds[(0 * 16 + nn) * LDSK + kh];
            const bf16_t* b1r  = &Wlds[(1 * 16 + nn) * LDSK + kh];
            const bf16_t* b2r  = &Wlds[(2 * 16 + nn) * LDSK + kh];
            const bf16_t* b3r  = &Wlds[(3 * 16 + nn) * LDSK + kh];
#pragma unroll 2
            for (int kb = 0; kb < K; kb += 32) {
                v16bf a = ld_frag(arow + kb, arow + kb + 16);
                v16bf b;
                b  = ld_frag(b0r + kb, b0r + kb + 16);
                a0 = __builtin_amdgcn_wmma_f32_16x16x32_bf16(
                         false, a, false, b, (short)0, a0, false, false);
                b  = ld_frag(b1r + kb, b1r + kb + 16);
                a1 = __builtin_amdgcn_wmma_f32_16x16x32_bf16(
                         false, a, false, b, (short)0, a1, false, false);
                b  = ld_frag(b2r + kb, b2r + kb + 16);
                a2 = __builtin_amdgcn_wmma_f32_16x16x32_bf16(
                         false, a, false, b, (short)0, a2, false, false);
                b  = ld_frag(b3r + kb, b3r + kb + 16);
                a3 = __builtin_amdgcn_wmma_f32_16x16x32_bf16(
                         false, a, false, b, (short)0, a3, false, false);
            }
            // ---- elementwise LSTM cell (wave-local, fp32, branch-free) ----
#pragma unroll
            for (int i = 0; i < 8; ++i) {
                float ig = sigmoidf_(a0[i] + bi);
                float fg = sigmoidf_(a1[i] + bff);
                float gg = tanhf_(a2[i] + bg);
                float og = sigmoidf_(a3[i] + bo);
                float cv = fg * creg[i] + ig * gg;
                creg[i]  = cv;
                float hv = og * tanhf_(cv);
                int m = wave * 16 + i + kh;          // C/D row mapping
                bf16_t hb = (bf16_t)hv;
                hcur[m * HDIM + wgbase + nn] = hb;
                outseq[((size_t)m * TLEN + t) * HDIM + wgbase + nn] = hb;
            }
        }
        grid_barrier(cnt, t + 1);                    // h_t visible to all WGs
    }
}

// ---------------- final FC + sigmoid ----------------
__global__ void fc_kernel(const bf16_t* __restrict__ h1seq,
                          const float* __restrict__ fc_w,
                          const float* __restrict__ fc_b,
                          float* __restrict__ out) {
    int b = threadIdx.x;                             // 64 threads
    if (b >= BATCH) return;
    const bf16_t* hrow = h1seq + ((size_t)b * TLEN + (TLEN - 1)) * HDIM;
    float s = fc_b[0];
    for (int j = 0; j < HDIM; ++j) s += (float)hrow[j] * fc_w[j];
    out[b] = sigmoidf_(s);
}

// ---------------- host side ----------------
extern "C" void kernel_launch(void* const* d_in, const int* in_sizes, int n_in,
                              void* d_out, int out_size, void* d_ws,
                              size_t ws_size, hipStream_t stream) {
    const float* x      = (const float*)d_in[0];
    const float* w_ih_0 = (const float*)d_in[1];
    const float* w_hh_0 = (const float*)d_in[2];
    const float* b_ih_0 = (const float*)d_in[3];
    const float* b_hh_0 = (const float*)d_in[4];
    const float* w_ih_1 = (const float*)d_in[5];
    const float* w_hh_1 = (const float*)d_in[6];
    const float* b_ih_1 = (const float*)d_in[7];
    const float* b_hh_1 = (const float*)d_in[8];
    const float* fc_w   = (const float*)d_in[9];
    const float* fc_b   = (const float*)d_in[10];

    // ---- workspace layout ----
    char* ws = (char*)d_ws;
    size_t off = 0;
    auto carve = [&](size_t bytes) -> char* {
        char* p = ws + off;
        off += (bytes + 255) & ~(size_t)255;
        return p;
    };
    const int K0 = INDIM + HDIM;   // 640
    const int K1 = HDIM + HDIM;    // 1024
    bf16_t* xbf    = (bf16_t*)carve((size_t)BATCH * TLEN * INDIM * 2);
    bf16_t* W0     = (bf16_t*)carve((size_t)G4H * K0 * 2);
    bf16_t* W1     = (bf16_t*)carve((size_t)G4H * K1 * 2);
    float*  bias0  = (float*)carve((size_t)G4H * 4);
    float*  bias1  = (float*)carve((size_t)G4H * 4);
    bf16_t* h0seq  = (bf16_t*)carve((size_t)BATCH * TLEN * HDIM * 2);
    bf16_t* h1seq  = (bf16_t*)carve((size_t)BATCH * TLEN * HDIM * 2);
    bf16_t* hcur   = (bf16_t*)carve((size_t)BATCH * HDIM * 2);
    int*    cnt    = (int*)carve((size_t)2 * (TLEN + 1) * 4);
    (void)ws_size; (void)in_sizes; (void)n_in; (void)out_size;

    // ---- allow >64KB dynamic LDS for the scan kernels ----
    const size_t lds0 = (size_t)2 * 64 * (K0 + LDS_PAD) * sizeof(bf16_t);
    const size_t lds1 = (size_t)2 * 64 * (K1 + LDS_PAD) * sizeof(bf16_t);
    hipFuncSetAttribute((const void*)lstm_scan_kernel<INDIM>,
                        hipFuncAttributeMaxDynamicSharedMemorySize, (int)lds0);
    hipFuncSetAttribute((const void*)lstm_scan_kernel<HDIM>,
                        hipFuncAttributeMaxDynamicSharedMemorySize, (int)lds1);

    // ---- precompute ----
    reset_kernel<<<4, 256, 0, stream>>>(cnt, 2 * (TLEN + 1));
    f32_to_bf16_kernel<<<4096, 256, 0, stream>>>(x, xbf,
                                                 BATCH * TLEN * INDIM);
    pack_w_kernel<<<2048, 256, 0, stream>>>(w_ih_0, w_hh_0, W0, INDIM);
    pack_w_kernel<<<2048, 256, 0, stream>>>(w_ih_1, w_hh_1, W1, HDIM);
    bias_kernel<<<(G4H + 255) / 256, 256, 0, stream>>>(b_ih_0, b_hh_0, bias0);
    bias_kernel<<<(G4H + 255) / 256, 256, 0, stream>>>(b_ih_1, b_hh_1, bias1);

    // ---- recurrent scans (persistent, 32 WGs each) ----
    lstm_scan_kernel<INDIM><<<NWG, SCAN_THREADS, lds0, stream>>>(
        xbf, W0, bias0, hcur, h0seq, cnt);
    lstm_scan_kernel<HDIM><<<NWG, SCAN_THREADS, lds1, stream>>>(
        h0seq, W1, bias1, hcur, h1seq, cnt + (TLEN + 1));

    // ---- classifier head ----
    fc_kernel<<<1, 64, 0, stream>>>(h1seq, fc_w, fc_b, (float*)d_out);
}
